// AttentionGAT_81355270521378
// MI455X (gfx1250) — matrix operand order repaired and checked by reading, not compiled, not run on hardware
//
#include <hip/hip_runtime.h>
#include <hip/hip_bf16.h>

// ---------------- problem constants ----------------
#define N_NODES   30000
#define N_EDGES   960000
#define E2        (N_EDGES + N_NODES)   // edges + self loops = 990000
#define IN_DIM    4527
#define KPAD      4544                  // IN_DIM rounded up to multiple of 32
#define HOG_DIM   4464
#define HID       128
#define F1        256                   // 2 heads * 128
#define NG        64
#define NC        4

// ---------------- WMMA types ----------------
typedef __attribute__((ext_vector_type(16))) __bf16 v16bf;
typedef __attribute__((ext_vector_type(8)))  float  v8f;

union FragB { unsigned u[8]; v16bf v; };

__device__ __forceinline__ v8f wmma_bf16(const FragB& a, const FragB& b, v8f c) {
  return __builtin_amdgcn_wmma_f32_16x16x32_bf16(false, a.v, false, b.v,
                                                 (short)0, c, false, false);
}

__device__ __forceinline__ unsigned short f2bf(float f) {
  unsigned u = __float_as_uint(f);
  unsigned r = (u + 0x7FFFu + ((u >> 16) & 1u)) >> 16;  // RNE
  return (unsigned short)r;
}

// order-preserving float <-> uint for atomicMax-based segment max
__device__ __forceinline__ unsigned fenc(float f) {
  unsigned u = __float_as_uint(f);
  return (u & 0x80000000u) ? ~u : (u | 0x80000000u);
}
__device__ __forceinline__ float fdec(unsigned u) {
  return (u & 0x80000000u) ? __uint_as_float(u ^ 0x80000000u)
                           : __uint_as_float(~u);
}

// 16-bit A 16x32 fragment: K = (lane/16)*16 + 2v+j (matches load_fragB).
// ldsA row stride = 36 elems.
__device__ __forceinline__ void load_fragA(const unsigned short* ldsA,
                                           int row, int lane, unsigned* u) {
  const int grp = (lane >> 4) << 4;
  const unsigned short* rp = ldsA + row * 36;
#pragma unroll
  for (int v = 0; v < 8; ++v) {
    int kk = (v << 1) + grp;
    u[v] = *(const unsigned*)(rp + kk);
  }
}

// 16-bit B 32x16 fragment: N = colBase + lane%16, K = (lane/16)*16 + 2v+j.
// ldsB is N-major with stride 34 elems (bank-conflict padding).
__device__ __forceinline__ void load_fragB(const unsigned short* ldsB,
                                           int colBase, int lane, unsigned* u) {
  const int n = colBase + (lane & 15);
  const int kofs = (lane >> 4) << 4;
  const unsigned short* cp = ldsB + n * 34 + kofs;
#pragma unroll
  for (int v = 0; v < 8; ++v) u[v] = *(const unsigned*)(cp + (v << 1));
}

// ---------------- TDM (Tensor Data Mover) support ----------------
#if defined(__has_builtin)
#if __has_builtin(__builtin_amdgcn_tensor_load_to_lds) && \
    __has_builtin(__builtin_amdgcn_s_wait_tensorcnt)
#define HAVE_TDM 1
#endif
#endif

#ifdef HAVE_TDM
typedef __attribute__((ext_vector_type(4))) unsigned tdm_v4u;
typedef __attribute__((ext_vector_type(8))) int      tdm_v8i;
typedef __attribute__((ext_vector_type(4))) int      tdm_v4i;

// Issue a TDM load of a [tileRows x tileCols] bf16 tile from a row-major
// tensor (row stride strideElems, remaining valid rows nRowsValid) at gaddr
// into LDS at lds_off, padded: after every 16 DWORDs (32 bf16) insert
// 2 DWORDs (4 bf16) -> LDS row stride 36 elements, matching load_fragA.
__device__ __forceinline__ void tdm_load_tile_padded(
    unsigned lds_off, unsigned long long gaddr, int nRowsValid,
    int tileRows, int tileCols, int strideElems) {
  tdm_v4u g0;
  g0[0] = 1u;                                  // count=1, load, user mode
  g0[1] = lds_off;                             // lds_addr
  g0[2] = (unsigned)(gaddr & 0xFFFFFFFFull);   // global_addr[31:0]
  g0[3] = (unsigned)((gaddr >> 32) & 0x01FFFFFFull) | 0x80000000u; // type=2

  unsigned dim1 = (unsigned)(nRowsValid < 0 ? 0 : nRowsValid);
  tdm_v8i g1;
  g1[0] = (int)((1u << 16)      // data_size = 2 bytes (bf16)
              | (1u << 20)      // pad_enable
              | (3u << 22)      // pad_interval: every 16 DWORDs
              | (1u << 25));    // pad_amount: 2 DWORDs
  g1[1] = (int)(((unsigned)strideElems & 0xFFFFu) << 16);  // tensor_dim0 lo16
  g1[2] = (int)((((unsigned)strideElems >> 16) & 0xFFFFu)  // tensor_dim0 hi16
              | ((dim1 & 0xFFFFu) << 16));                 // tensor_dim1 lo16
  g1[3] = (int)(((dim1 >> 16) & 0xFFFFu)                   // tensor_dim1 hi16
              | (((unsigned)tileCols & 0xFFFFu) << 16));   // tile_dim0
  g1[4] = (int)((unsigned)tileRows & 0xFFFFu);             // tile_dim1
  g1[5] = (int)(unsigned)strideElems;                      // tensor_dim0_stride lo32
  g1[6] = 0;
  g1[7] = 0;

  tdm_v4i g2; g2[0] = 0; g2[1] = 0; g2[2] = 0; g2[3] = 0;
  tdm_v4i g3; g3[0] = 0; g3[1] = 0; g3[2] = 0; g3[3] = 0;
  tdm_v8i g4; g4[0] = 0; g4[1] = 0; g4[2] = 0; g4[3] = 0;
  g4[4] = 0; g4[5] = 0; g4[6] = 0; g4[7] = 0;

  // 6-arg form (amdgpu-toolchain / therock-10.0 headers)
  __builtin_amdgcn_tensor_load_to_lds(g0, g1, g2, g3, g4, 0);
}
#endif

// ---------------- utility fill ----------------
__global__ void k_fill(unsigned* __restrict__ p, unsigned v, long n) {
  long i = (long)blockIdx.x * blockDim.x + threadIdx.x;
  long stride = (long)gridDim.x * blockDim.x;
  for (; i < n; i += stride) p[i] = v;
}

// ---------------- 1) per-node attention softmax ----------------
__global__ __launch_bounds__(128) void k_attn(const float* __restrict__ x,
                                              const float* __restrict__ aw,
                                              const float* __restrict__ ab,
                                              float* __restrict__ attn) {
  int node = blockIdx.x * 4 + (threadIdx.x >> 5);
  int lane = threadIdx.x & 31;
  if (node >= N_NODES) return;
  const float* xr = x + (long)node * IN_DIM;
  float s0 = 0.f, s1 = 0.f;
  for (int i = lane; i < IN_DIM; i += 32) {
    float xv = xr[i];
    s0 += xv * aw[i * 2 + 0];
    s1 += xv * aw[i * 2 + 1];
  }
#pragma unroll
  for (int o = 16; o > 0; o >>= 1) {
    s0 += __shfl_xor(s0, o, 32);
    s1 += __shfl_xor(s1, o, 32);
  }
  if (lane == 0) {
    float l0 = s0 + ab[0], l1 = s1 + ab[1];
    float mx = fmaxf(l0, l1);
    float e0 = __expf(l0 - mx), e1 = __expf(l1 - mx);
    float d = e0 + e1;
    attn[node * 2 + 0] = e0 / d;
    attn[node * 2 + 1] = e1 / d;
  }
}

// ---------------- weight f32 -> bf16 (with zero K padding) ----------------
__global__ void k_convW(const float* __restrict__ W, unsigned short* __restrict__ Wbf,
                        int rows_valid, long total, int cols) {
  long i = (long)blockIdx.x * blockDim.x + threadIdx.x;
  long stride = (long)gridDim.x * blockDim.x;
  for (; i < total; i += stride) {
    int r = (int)(i / cols);
    float v = (r < rows_valid) ? W[i] : 0.f;
    Wbf[i] = f2bf(v);
  }
}

// ---------------- 2) GEMM1: (scaled x)[30000 x 4544] @ W1bf[4544 x 256] ----------------
__global__ __launch_bounds__(256) void k_gemm1(const float* __restrict__ x,
                                               const float* __restrict__ attn,
                                               const unsigned short* __restrict__ Bw,
                                               float* __restrict__ C) {
  __shared__ unsigned short ldsA[64 * 36];
  __shared__ unsigned short ldsB[256 * 34];
  const int tid = threadIdx.x;
  const int lane = tid & 31, wave = tid >> 5;
  const int wm = wave & 3, wn = wave >> 2;
  const int rowBase = blockIdx.x * 64;

  v8f acc[8];
#pragma unroll
  for (int i = 0; i < 8; ++i)
#pragma unroll
    for (int r = 0; r < 8; ++r) acc[i][r] = 0.f;

  const int arow = tid >> 2;         // 0..63
  const int acol0 = (tid & 3) * 8;   // 0,8,16,24
  const int grow = rowBase + arow;
  float sc0 = 0.f, sc1 = 0.f;
  if (grow < N_NODES) { sc0 = attn[grow * 2 + 0]; sc1 = attn[grow * 2 + 1]; }

  for (int k0 = 0; k0 < KPAD; k0 += 32) {
    // stage A tile 64x32 (scale + bf16 convert on the fly)
    {
      unsigned short* dp = ldsA + arow * 36 + acol0;
      // prefetch next K-tile of the x stream (global_prefetch_b8)
      if (grow < N_NODES && (k0 + 32 + acol0) < IN_DIM)
        __builtin_prefetch(&x[(long)grow * IN_DIM + k0 + 32 + acol0], 0, 3);
#pragma unroll
      for (int j = 0; j < 8; ++j) {
        int kg = k0 + acol0 + j;
        float v = 0.f;
        if (grow < N_NODES && kg < IN_DIM) {
          float s = (kg < HOG_DIM) ? sc0 : sc1;
          v = s * x[(long)grow * IN_DIM + kg];
        }
        dp[j] = f2bf(v);
      }
    }
    // stage B tile 32x256 transposed (N-major)
    for (int idx = tid; idx < 32 * 256; idx += 256) {
      int kk = idx >> 8;
      int nn = idx & 255;
      ldsB[nn * 34 + kk] = Bw[(long)(k0 + kk) * 256 + nn];
    }
    __syncthreads();

    FragB fa;
    load_fragA(ldsA, wm * 16 + (lane & 15), lane, fa.u);
#pragma unroll
    for (int nt = 0; nt < 8; ++nt) {
      FragB fb;
      load_fragB(ldsB, wn * 128 + nt * 16, lane, fb.u);
      acc[nt] = wmma_bf16(fa, fb, acc[nt]);
    }
    __syncthreads();
  }

#pragma unroll
  for (int nt = 0; nt < 8; ++nt) {
    int col = wn * 128 + nt * 16 + (lane & 15);
#pragma unroll
    for (int r = 0; r < 8; ++r) {
      int gr = rowBase + wm * 16 + r + ((lane >> 4) << 3);
      if (gr < N_NODES) C[(long)gr * F1 + col] = acc[nt][r];
    }
  }
}

// ---------------- 6) GEMM2: h1bf[30000 x 256] @ W2bf[256 x 128] ----------------
// A tile is staged by the Tensor Data Mover with LDS padding producing the
// 36-element row stride load_fragA expects; TENSORcnt synchronizes.
__global__ __launch_bounds__(256) void k_gemm2(const unsigned short* __restrict__ A,
                                               const unsigned short* __restrict__ Bw,
                                               float* __restrict__ C) {
  __shared__ unsigned short ldsA[64 * 36];
  __shared__ unsigned short ldsB[128 * 34];
  const int tid = threadIdx.x;
  const int lane = tid & 31, wave = tid >> 5;
  const int wm = wave & 3, wn = wave >> 2;
  const int rowBase = blockIdx.x * 64;

  v8f acc[4];
#pragma unroll
  for (int i = 0; i < 4; ++i)
#pragma unroll
    for (int r = 0; r < 8; ++r) acc[i][r] = 0.f;

#ifdef HAVE_TDM
  const unsigned ldsA_off = (unsigned)(size_t)(void*)ldsA;
  const unsigned long long gaddr0 =
      (unsigned long long)(size_t)(const void*)(A + (long)rowBase * F1);
  const int remRows = N_NODES - rowBase;
#else
  const int arow = tid >> 2;
  const int acol0 = (tid & 3) * 8;
  const int grow = rowBase + arow;
#endif

  for (int k0 = 0; k0 < F1; k0 += 32) {
#ifdef HAVE_TDM
    if (wave == 0) {
      // 64x32 bf16 tile at column k0; zero-fill beyond remRows rows.
      tdm_load_tile_padded(ldsA_off, gaddr0 + (unsigned long long)k0 * 2,
                           remRows, 64, 32, F1);
    }
#else
    {
      unsigned short* dp = ldsA + arow * 36 + acol0;
#pragma unroll
      for (int j = 0; j < 8; ++j)
        dp[j] = (grow < N_NODES) ? A[(long)grow * F1 + k0 + acol0 + j]
                                 : (unsigned short)0;
    }
#endif
    for (int idx = tid; idx < 32 * 128; idx += 256) {
      int kk = idx >> 7;
      int nn = idx & 127;
      ldsB[nn * 34 + kk] = Bw[(long)(k0 + kk) * HID + nn];
    }
#ifdef HAVE_TDM
    if (wave == 0) __builtin_amdgcn_s_wait_tensorcnt(0);
#endif
    __syncthreads();

    FragB fa;
    load_fragA(ldsA, wm * 16 + (lane & 15), lane, fa.u);
#pragma unroll
    for (int nt = 0; nt < 4; ++nt) {
      FragB fb;
      load_fragB(ldsB, wn * 64 + nt * 16, lane, fb.u);
      acc[nt] = wmma_bf16(fa, fb, acc[nt]);
    }
    __syncthreads();
  }

#pragma unroll
  for (int nt = 0; nt < 4; ++nt) {
    int col = wn * 64 + nt * 16 + (lane & 15);
#pragma unroll
    for (int r = 0; r < 8; ++r) {
      int gr = rowBase + wm * 16 + r + ((lane >> 4) << 3);
      if (gr < N_NODES) C[(long)gr * HID + col] = acc[nt][r];
    }
  }
}

// ---------------- per-(node,head) attention logits alpha_s / alpha_d ----------------
__global__ __launch_bounds__(128) void k_nodealpha(const float* __restrict__ h,
                                                   const float* __restrict__ asrc,
                                                   const float* __restrict__ adst,
                                                   float* __restrict__ as_out,
                                                   float* __restrict__ ad_out,
                                                   int H) {
  int item = blockIdx.x * 4 + (threadIdx.x >> 5);
  int lane = threadIdx.x & 31;
  if (item >= N_NODES * H) return;
  int node = item / H;
  int hh = item - node * H;
  int F = H * HID;
  const float* hr = h + (long)node * F + hh * HID;
  const float* sr = asrc + hh * HID;
  const float* dr = adst + hh * HID;
  float ss = 0.f, sd = 0.f;
#pragma unroll
  for (int i = 0; i < HID / 32; ++i) {
    int d = lane + 32 * i;
    float hv = hr[d];
    ss += hv * sr[d];
    sd += hv * dr[d];
  }
#pragma unroll
  for (int o = 16; o > 0; o >>= 1) {
    ss += __shfl_xor(ss, o, 32);
    sd += __shfl_xor(sd, o, 32);
  }
  if (lane == 0) { as_out[item] = ss; ad_out[item] = sd; }
}

__device__ __forceinline__ void edge_sd(const int* __restrict__ ei, int e,
                                        int& s, int& d) {
  if (e < N_EDGES) { s = ei[e]; d = ei[N_EDGES + e]; }
  else             { s = d = e - N_EDGES; }
}

// ---------------- edge pass A: segment max (via uint-encoded atomicMax) ----------------
__global__ void k_edge_max(const int* __restrict__ ei, const float* __restrict__ as,
                           const float* __restrict__ ad, unsigned* __restrict__ m,
                           int H) {
  long tot = (long)E2 * H;
  long i = (long)blockIdx.x * blockDim.x + threadIdx.x;
  long stride = (long)gridDim.x * blockDim.x;
  for (; i < tot; i += stride) {
    int e = (int)(i / H), hh = (int)(i - (long)e * H);
    int s, d; edge_sd(ei, e, s, d);
    float ev = as[s * H + hh] + ad[d * H + hh];
    ev = (ev > 0.f) ? ev : 0.2f * ev;          // leaky_relu 0.2
    atomicMax(&m[d * H + hh], fenc(ev));
  }
}

// ---------------- edge pass B: exp(e - max) and segment sum ----------------
__global__ void k_edge_exp(const int* __restrict__ ei, const float* __restrict__ as,
                           const float* __restrict__ ad, const unsigned* __restrict__ m,
                           float* __restrict__ den, float* __restrict__ ee, int H) {
  long tot = (long)E2 * H;
  long i = (long)blockIdx.x * blockDim.x + threadIdx.x;
  long stride = (long)gridDim.x * blockDim.x;
  for (; i < tot; i += stride) {
    int e = (int)(i / H), hh = (int)(i - (long)e * H);
    int s, d; edge_sd(ei, e, s, d);
    float ev = as[s * H + hh] + ad[d * H + hh];
    ev = (ev > 0.f) ? ev : 0.2f * ev;
    float ex = __expf(ev - fdec(m[d * H + hh]));
    ee[i] = ex;
    atomicAdd(&den[d * H + hh], ex);
  }
}

// ---------------- edge pass C: normalize (in place over ee) ----------------
__global__ void k_edge_alpha(const int* __restrict__ ei, float* __restrict__ ee,
                             const float* __restrict__ den, int H) {
  long tot = (long)E2 * H;
  long i = (long)blockIdx.x * blockDim.x + threadIdx.x;
  long stride = (long)gridDim.x * blockDim.x;
  for (; i < tot; i += stride) {
    int e = (int)(i / H), hh = (int)(i - (long)e * H);
    int s, d; edge_sd(ei, e, s, d);
    (void)s;
    ee[i] = ee[i] / (den[d * H + hh] + 1e-16f);
  }
}

// ---------------- edge pass D: scatter-aggregate messages ----------------
__global__ void k_edge_agg(const int* __restrict__ ei, const float* __restrict__ alpha,
                           const float* __restrict__ h, float* __restrict__ out,
                           int H) {
  int e = blockIdx.x;
  int t = threadIdx.x;           // blockDim = H*128
  int F = H * HID;
  int hh = t >> 7;
  int s, d; edge_sd(ei, e, s, d);
  float al = alpha[(long)e * H + hh];
  atomicAdd(&out[(long)d * F + t], al * h[(long)s * F + t]);
}

// ---------------- bias + relu + bf16 convert ----------------
__global__ void k_biasrelu_bf(const float* __restrict__ in, const float* __restrict__ bias,
                              unsigned short* __restrict__ obf, long n, int F) {
  long i = (long)blockIdx.x * blockDim.x + threadIdx.x;
  long stride = (long)gridDim.x * blockDim.x;
  for (; i < n; i += stride) {
    float v = in[i] + bias[i % F];
    obf[i] = f2bf(fmaxf(v, 0.f));
  }
}

// ---------------- pooling: bias + relu + per-graph sums/counts ----------------
__global__ __launch_bounds__(128) void k_pool(const float* __restrict__ out2,
                                              const float* __restrict__ b2,
                                              const int* __restrict__ batch,
                                              float* __restrict__ pool,
                                              float* __restrict__ cnt) {
  int n = blockIdx.x;
  int d = threadIdx.x;
  if (n >= N_NODES) return;
  float v = fmaxf(out2[(long)n * HID + d] + b2[d], 0.f);
  int g = batch[n];
  atomicAdd(&pool[g * HID + d], v);
  if (d == 0) atomicAdd(&cnt[g], 1.0f);
}

// ---------------- classifier head (single block) ----------------
__global__ __launch_bounds__(256) void k_head(const float* __restrict__ pool,
                                              const float* __restrict__ cnt,
                                              const float* __restrict__ Wc1,
                                              const float* __restrict__ bc1,
                                              const float* __restrict__ Wc2,
                                              const float* __restrict__ bc2,
                                              float* __restrict__ out) {
  __shared__ float p[NG * HID];   // 32 KB
  __shared__ float z[NG * 64];    // 16 KB
  int t = threadIdx.x;
  for (int i = t; i < NG * HID; i += 256) {
    int g = i >> 7;
    p[i] = pool[i] / fmaxf(cnt[g], 1.0f);
  }
  __syncthreads();
  for (int i = t; i < NG * 64; i += 256) {
    int g = i >> 6, j = i & 63;
    float s = bc1[j];
    const float* pg = p + g * HID;
    for (int d = 0; d < HID; ++d) s += pg[d] * Wc1[d * 64 + j];
    z[i] = fmaxf(s, 0.f);
  }
  __syncthreads();
  if (t < NG * NC) {
    int g = t >> 2, c = t & 3;
    float s = bc2[c];
    const float* zg = z + g * 64;
    for (int j = 0; j < 64; ++j) s += zg[j] * Wc2[j * NC + c];
    out[t] = s;
  }
}

// ---------------- launch ----------------
extern "C" void kernel_launch(void* const* d_in, const int* in_sizes, int n_in,
                              void* d_out, int out_size, void* d_ws, size_t ws_size,
                              hipStream_t stream) {
  (void)in_sizes; (void)n_in; (void)out_size; (void)ws_size;
  const float* x     = (const float*)d_in[0];
  const int*   ei    = (const int*)d_in[1];
  const int*   batch = (const int*)d_in[2];
  const float* aw    = (const float*)d_in[3];
  const float* ab    = (const float*)d_in[4];
  const float* W1    = (const float*)d_in[5];
  const float* as1w  = (const float*)d_in[6];
  const float* ad1w  = (const float*)d_in[7];
  const float* b1    = (const float*)d_in[8];
  const float* W2    = (const float*)d_in[9];
  const float* as2w  = (const float*)d_in[10];
  const float* ad2w  = (const float*)d_in[11];
  const float* b2    = (const float*)d_in[12];
  const float* Wc1   = (const float*)d_in[13];
  const float* bc1   = (const float*)d_in[14];
  const float* Wc2   = (const float*)d_in[15];
  const float* bc2   = (const float*)d_in[16];

  // bump allocator over d_ws (256B aligned)
  char* base = (char*)d_ws;
  size_t off = 0;
  auto alloc = [&](size_t bytes) -> void* {
    off = (off + 255) & ~(size_t)255;
    void* p = base + off;
    off += bytes;
    return p;
  };

  float*          attn  = (float*)alloc((size_t)N_NODES * 2 * 4);
  unsigned short* W1bf  = (unsigned short*)alloc((size_t)KPAD * F1 * 2);
  float*          h1    = (float*)alloc((size_t)N_NODES * F1 * 4);
  float*          as1   = (float*)alloc((size_t)N_NODES * 2 * 4);
  float*          ad1   = (float*)alloc((size_t)N_NODES * 2 * 4);
  unsigned*       m1    = (unsigned*)alloc((size_t)N_NODES * 2 * 4);
  float*          den1  = (float*)alloc((size_t)N_NODES * 2 * 4);
  float*          ee1   = (float*)alloc((size_t)E2 * 2 * 4);
  float*          out1  = (float*)alloc((size_t)N_NODES * F1 * 4);
  unsigned short* h1bf  = (unsigned short*)alloc((size_t)N_NODES * F1 * 2);
  unsigned short* W2bf  = (unsigned short*)alloc((size_t)F1 * HID * 2);
  float*          h2    = (float*)alloc((size_t)N_NODES * HID * 4);
  float*          as2   = (float*)alloc((size_t)N_NODES * 4);
  float*          ad2   = (float*)alloc((size_t)N_NODES * 4);
  unsigned*       m2    = (unsigned*)alloc((size_t)N_NODES * 4);
  float*          den2  = (float*)alloc((size_t)N_NODES * 4);
  float*          ee2   = (float*)alloc((size_t)E2 * 4);
  float*          out2  = (float*)alloc((size_t)N_NODES * HID * 4);
  float*          pool  = (float*)alloc((size_t)NG * HID * 4);
  float*          cnt   = (float*)alloc((size_t)NG * 4);

  dim3 b256(256), b128(128);
  auto fill = [&](void* p, unsigned v, long n) {
    int g = (int)((n + 255) / 256);
    if (g > 4096) g = 4096;
    k_fill<<<dim3(g), b256, 0, stream>>>((unsigned*)p, v, n);
  };

  // zero accumulators / encoded -inf maxima (encoded 0 == minimum)
  fill(m1, 0u, (long)N_NODES * 2);
  fill(den1, 0u, (long)N_NODES * 2);
  fill(out1, 0u, (long)N_NODES * F1);
  fill(m2, 0u, N_NODES);
  fill(den2, 0u, N_NODES);
  fill(out2, 0u, (long)N_NODES * HID);
  fill(pool, 0u, (long)NG * HID);
  fill(cnt, 0u, NG);

  // (1) node attention
  k_attn<<<dim3((N_NODES + 3) / 4), b128, 0, stream>>>(x, aw, ab, attn);

  // weight conversion
  k_convW<<<dim3(2048), b256, 0, stream>>>(W1, W1bf, IN_DIM, (long)KPAD * F1, F1);
  k_convW<<<dim3(128), b256, 0, stream>>>(W2, W2bf, F1, (long)F1 * HID, HID);

  // (2) layer-1 projection GEMM (WMMA bf16)
  k_gemm1<<<dim3((N_NODES + 63) / 64), b256, 0, stream>>>(x, attn, W1bf, h1);

  // (3) layer-1 edge softmax + aggregation (H = 2)
  k_nodealpha<<<dim3((N_NODES * 2 + 3) / 4), b128, 0, stream>>>(h1, as1w, ad1w, as1, ad1, 2);
  k_edge_max<<<dim3(2048), b256, 0, stream>>>(ei, as1, ad1, m1, 2);
  k_edge_exp<<<dim3(2048), b256, 0, stream>>>(ei, as1, ad1, m1, den1, ee1, 2);
  k_edge_alpha<<<dim3(2048), b256, 0, stream>>>(ei, ee1, den1, 2);
  k_edge_agg<<<dim3(E2), b256, 0, stream>>>(ei, ee1, h1, out1, 2);

  // bias + relu + bf16 for layer-2 GEMM input
  k_biasrelu_bf<<<dim3(4096), b256, 0, stream>>>(out1, b1, h1bf, (long)N_NODES * F1, F1);

  // (4) layer-2 projection GEMM (WMMA bf16, TDM-staged A tiles)
  k_gemm2<<<dim3((N_NODES + 63) / 64), b256, 0, stream>>>(h1bf, W2bf, h2);

  // (5) layer-2 edge softmax + aggregation (H = 1)
  k_nodealpha<<<dim3((N_NODES + 3) / 4), b128, 0, stream>>>(h2, as2w, ad2w, as2, ad2, 1);
  k_edge_max<<<dim3(2048), b256, 0, stream>>>(ei, as2, ad2, m2, 1);
  k_edge_exp<<<dim3(2048), b256, 0, stream>>>(ei, as2, ad2, m2, den2, ee2, 1);
  k_edge_alpha<<<dim3(2048), b256, 0, stream>>>(ei, ee2, den2, 1);
  k_edge_agg<<<dim3(E2), b128, 0, stream>>>(ei, ee2, h2, out2, 1);

  // (6) pooling + classifier head
  k_pool<<<dim3(N_NODES), b128, 0, stream>>>(out2, b2, batch, pool, cnt);
  k_head<<<dim3(1), b256, 0, stream>>>(pool, cnt, Wc1, bc1, Wc2, bc2, (float*)d_out);
}
